// CFConv_mine_70944269795976
// MI455X (gfx1250) — matrix operand order repaired and checked by reading, not compiled, not run on hardware
//
#include <hip/hip_runtime.h>

// ---------------- problem constants ----------------
#define N_NODES 50000
#define N_EDGES 800000
#define CH      128      // IN_CH == FILTERS == OUT_CH
#define NG      50       // gaussians
#define NGP     64       // padded K for edge_attr GEMM
#define CUTOFF_F 10.0f
#define LOG2_F   0.6931471805599453f
#define PI_F     3.14159265358979323846f

typedef __attribute__((ext_vector_type(16))) __bf16 v16bf;
typedef __attribute__((ext_vector_type(8)))  float  v8f;

// ---------------- WMMA helpers ----------------
__device__ __forceinline__ v8f wmma_bf16(v16bf a, v16bf b, v8f c) {
  // D(16x16,f32) = A(16x32,bf16) x B(32x16,bf16) + C
  return __builtin_amdgcn_wmma_f32_16x16x32_bf16(
      /*neg_a=*/false, a, /*neg_b=*/false, b,
      /*c_mod=*/(short)0, c, /*reuse_a=*/false, /*reuse_b=*/false);
}

__device__ __forceinline__ v8f v8f_zero() {
  v8f z = {0.f, 0.f, 0.f, 0.f, 0.f, 0.f, 0.f, 0.f};
  return z;
}

// K index pattern for 16-bit A fragment (ISA 7.12.2):
// lanes 0-15: K = kb+{0..7} and kb+16+{0..7} with kb=k0
// lanes 16-31: same with kb=k0+8
__device__ __forceinline__ int a_koff(int i) {
  return (i < 4) ? (2 * i) : (16 + 2 * (i - 4));
}

// A fragment from row-major f32 source (convert to bf16 on the fly)
__device__ __forceinline__ v16bf load_a_f32(const float* __restrict__ src, int lda,
                                            int lane, int k0) {
  int r  = lane & 15;
  int kb = k0 + ((lane >> 4) ? 8 : 0);
  const float* p = src + (size_t)r * lda + kb;
  v16bf a;
#pragma unroll
  for (int i = 0; i < 8; ++i) {
    int k = a_koff(i);
    a[2 * i]     = (__bf16)p[k];
    a[2 * i + 1] = (__bf16)p[k + 1];
  }
  return a;
}

// A fragment = bf16(x + y), both row-major f32
__device__ __forceinline__ v16bf load_a_sum(const float* __restrict__ x,
                                            const float* __restrict__ y,
                                            int lda, int lane, int k0) {
  int r  = lane & 15;
  int kb = k0 + ((lane >> 4) ? 8 : 0);
  const float* px = x + (size_t)r * lda + kb;
  const float* py = y + (size_t)r * lda + kb;
  v16bf a;
#pragma unroll
  for (int i = 0; i < 8; ++i) {
    int k = a_koff(i);
    a[2 * i]     = (__bf16)(px[k] + py[k]);
    a[2 * i + 1] = (__bf16)(px[k + 1] + py[k + 1]);
  }
  return a;
}

// A fragment from edge_attr (K padded 50 -> 64 with zeros)
__device__ __forceinline__ v16bf load_a_attr(const float* __restrict__ attr,
                                             int e0, int lane, int k0) {
  int r  = lane & 15;
  int kb = k0 + ((lane >> 4) ? 8 : 0);
  const float* p = attr + (size_t)(e0 + r) * NG;
  v16bf a;
#pragma unroll
  for (int i = 0; i < 8; ++i) {
    int k = kb + a_koff(i);
    a[2 * i]     = (k < NG)     ? (__bf16)p[k]     : (__bf16)0.0f;
    a[2 * i + 1] = (k + 1 < NG) ? (__bf16)p[k + 1] : (__bf16)0.0f;
  }
  return a;
}

// A fragment from bf16 LDS tile (row-major, stride lda)
__device__ __forceinline__ v16bf load_a_bf16(const __bf16* src, int lda,
                                             int lane, int k0) {
  int r  = lane & 15;
  int kb = k0 + ((lane >> 4) ? 8 : 0);
  const __bf16* p = src + r * lda + kb;
  v16bf a;
#pragma unroll
  for (int i = 0; i < 8; ++i) {
    int k = a_koff(i);
    a[2 * i]     = p[k];
    a[2 * i + 1] = p[k + 1];
  }
  return a;
}

// B fragment (32x16, KxN) from row-major weight W[N][K] (B[k][n] = W[n][k]).
// lane n = lane&15, K = kb+{0..7} and kb+16+{0..7} (kb = 0 or 8 by lane half).
__device__ __forceinline__ v16bf load_b(const __bf16* __restrict__ W, int ldw, int lane) {
  int n  = lane & 15;
  int kb = (lane >> 4) ? 8 : 0;
  const __bf16* p = W + (size_t)n * ldw + kb;
  v16bf b;
#pragma unroll
  for (int i = 0; i < 16; ++i) b[i] = p[i < 8 ? i : i + 8];
  return b;
}

// shifted softplus: softplus(x) - log(2), numerically stable
__device__ __forceinline__ float ssp(float x) {
  return fmaxf(x, 0.0f) + log1pf(__expf(-fabsf(x))) - LOG2_F;
}

__device__ __forceinline__ void atomic_add_f32(float* p, float v) {
  unsafeAtomicAdd(p, v);   // hardware global_atomic_add_f32
}

// ---------------- kernel 1: weight conversion (f32 -> bf16, pad nn_w1) ----------------
__global__ void __launch_bounds__(256)
k_convert(const float* __restrict__ lin1, const float* __restrict__ w1,
          const float* __restrict__ w2,   const float* __restrict__ lin2,
          unsigned short* lin1b, unsigned short* w1b,
          unsigned short* w2b,   unsigned short* lin2b) {
  int i = blockIdx.x * blockDim.x + threadIdx.x;
  __bf16* L1 = (__bf16*)lin1b;
  __bf16* W1 = (__bf16*)w1b;
  __bf16* W2 = (__bf16*)w2b;
  __bf16* L2 = (__bf16*)lin2b;
  if (i < CH * CH) {
    L1[i] = (__bf16)lin1[i];
    W2[i] = (__bf16)w2[i];
    L2[i] = (__bf16)lin2[i];
  }
  if (i < CH * NGP) {
    int r = i >> 6, c = i & 63;
    W1[i] = (c < NG) ? (__bf16)w1[r * NG + c] : (__bf16)0.0f;
  }
}

// ---------------- kernel 2: h1 = h @ lin1_w^T  (WMMA) ----------------
__global__ void __launch_bounds__(128)
k_node_gemm(const float* __restrict__ h, const unsigned short* __restrict__ w_us,
            float* __restrict__ h1) {
  const __bf16* W = (const __bf16*)w_us;
  int lane = threadIdx.x & 31;
  int wave = threadIdx.x >> 5;
  int m0 = (blockIdx.x * 4 + wave) * 16;
  if (m0 >= N_NODES) return;

  v8f acc[8];
#pragma unroll
  for (int f = 0; f < 8; ++f) acc[f] = v8f_zero();

  const float* A = h + (size_t)m0 * CH;
#pragma unroll
  for (int kk = 0; kk < CH; kk += 32) {
    v16bf a = load_a_f32(A, CH, lane, kk);
#pragma unroll
    for (int f = 0; f < 8; ++f) {
      v16bf b = load_b(W + (size_t)(f * 16) * CH + kk, CH, lane);
      acc[f] = wmma_bf16(a, b, acc[f]);
    }
  }
  int col = lane & 15, hi = lane >> 4;
#pragma unroll
  for (int f = 0; f < 8; ++f)
#pragma unroll
    for (int v = 0; v < 8; ++v)
      h1[(size_t)(m0 + v + 8 * hi) * CH + f * 16 + col] = acc[f][v];
}

// ---------------- kernel 3: fused edge MLP + gather/scatter ----------------
// per wave: one 16-edge tile.  GEMM1 (K=64) -> ssp -> LDS bf16 ->
// GEMM2 (K=128) -> ssp -> LDS f32 -> cutoff * gather(h1[src]) -> atomic agg[dst]
__global__ void __launch_bounds__(128)
k_edge(const float* __restrict__ eattr, const float* __restrict__ ew,
       const int* __restrict__ eidx,
       const unsigned short* __restrict__ w1_us, const float* __restrict__ b1,
       const unsigned short* __restrict__ w2_us, const float* __restrict__ b2,
       const float* __restrict__ h1, float* __restrict__ agg) {
  const __bf16* W1 = (const __bf16*)w1_us;
  const __bf16* W2 = (const __bf16*)w2_us;
  __shared__ __bf16 xsh[4][16][CH];   // 16 KB: GEMM1 output as bf16 (A of GEMM2)
  __shared__ float  wsh[4][16][CH];   // 32 KB: final filter values

  int lane = threadIdx.x & 31;
  int wave = threadIdx.x >> 5;
  int e0 = (blockIdx.x * 4 + wave) * 16;
  if (e0 >= N_EDGES) return;
  int col = lane & 15, hi = lane >> 4;

  // ---- GEMM1: x = ssp(edge_attr @ w1^T + b1), K padded to 64
  v8f acc[8];
#pragma unroll
  for (int f = 0; f < 8; ++f) acc[f] = v8f_zero();
#pragma unroll
  for (int kk = 0; kk < NGP; kk += 32) {
    v16bf a = load_a_attr(eattr, e0, lane, kk);
#pragma unroll
    for (int f = 0; f < 8; ++f) {
      v16bf b = load_b(W1 + (size_t)(f * 16) * NGP + kk, NGP, lane);
      acc[f] = wmma_bf16(a, b, acc[f]);
    }
  }
#pragma unroll
  for (int f = 0; f < 8; ++f) {
    int n = f * 16 + col;
    float bias = b1[n];
#pragma unroll
    for (int v = 0; v < 8; ++v)
      xsh[wave][v + 8 * hi][n] = (__bf16)ssp(acc[f][v] + bias);
  }
  // wave-local LDS transpose: wait for our DS stores before cross-lane reads
  asm volatile("s_wait_dscnt 0" ::: "memory");

  // ---- GEMM2: W = ssp(x @ w2^T + b2)
#pragma unroll
  for (int f = 0; f < 8; ++f) acc[f] = v8f_zero();
#pragma unroll
  for (int kk = 0; kk < CH; kk += 32) {
    v16bf a = load_a_bf16(&xsh[wave][0][0], CH, lane, kk);
#pragma unroll
    for (int f = 0; f < 8; ++f) {
      v16bf b = load_b(W2 + (size_t)(f * 16) * CH + kk, CH, lane);
      acc[f] = wmma_bf16(a, b, acc[f]);
    }
  }
#pragma unroll
  for (int f = 0; f < 8; ++f) {
    int n = f * 16 + col;
    float bias = b2[n];
#pragma unroll
    for (int v = 0; v < 8; ++v)
      wsh[wave][v + 8 * hi][n] = ssp(acc[f][v] + bias);
  }
  asm volatile("s_wait_dscnt 0" ::: "memory");

  // ---- cutoff scale, gather h1[src], scatter-add to agg[dst]
  // h1/agg are L2-resident (25.6 MB each << 192 MB L2); channel-coalesced lanes.
#pragma unroll 1
  for (int el = 0; el < 16; ++el) {
    int e   = e0 + el;
    int dst = eidx[e];            // row 0: segment (destination) ids
    int src = eidx[N_EDGES + e];  // row 1: source (gather) ids
    float ce = 0.5f * (__cosf(ew[e] * (PI_F / CUTOFF_F)) + 1.0f);
    const float* hs = h1 + (size_t)src * CH;
    float* ad = agg + (size_t)dst * CH;
#pragma unroll
    for (int j = 0; j < 4; ++j) {
      int ch = lane + 32 * j;
      float val = wsh[wave][el][ch] * ce * hs[ch];
      atomic_add_f32(ad + ch, val);
    }
  }
}

// ---------------- kernel 4: out = (h1 + agg) @ lin2_w^T + lin2_b ----------------
__global__ void __launch_bounds__(128)
k_out_gemm(const float* __restrict__ h1, const float* __restrict__ agg,
           const unsigned short* __restrict__ w_us, const float* __restrict__ bias,
           float* __restrict__ out) {
  const __bf16* W = (const __bf16*)w_us;
  int lane = threadIdx.x & 31;
  int wave = threadIdx.x >> 5;
  int m0 = (blockIdx.x * 4 + wave) * 16;
  if (m0 >= N_NODES) return;

  v8f acc[8];
#pragma unroll
  for (int f = 0; f < 8; ++f) acc[f] = v8f_zero();

  const float* X = h1 + (size_t)m0 * CH;
  const float* Y = agg + (size_t)m0 * CH;
#pragma unroll
  for (int kk = 0; kk < CH; kk += 32) {
    v16bf a = load_a_sum(X, Y, CH, lane, kk);
#pragma unroll
    for (int f = 0; f < 8; ++f) {
      v16bf b = load_b(W + (size_t)(f * 16) * CH + kk, CH, lane);
      acc[f] = wmma_bf16(a, b, acc[f]);
    }
  }
  int col = lane & 15, hi = lane >> 4;
#pragma unroll
  for (int f = 0; f < 8; ++f) {
    int n = f * 16 + col;
    float bn = bias[n];
#pragma unroll
    for (int v = 0; v < 8; ++v)
      out[(size_t)(m0 + v + 8 * hi) * CH + n] = acc[f][v] + bn;
  }
}

// ---------------- launcher ----------------
extern "C" void kernel_launch(void* const* d_in, const int* in_sizes, int n_in,
                              void* d_out, int out_size, void* d_ws, size_t ws_size,
                              hipStream_t stream) {
  const float* h       = (const float*)d_in[0];
  const int*   eidx    = (const int*)d_in[1];
  const float* ew      = (const float*)d_in[2];
  const float* eattr   = (const float*)d_in[3];
  const float* lin1_w  = (const float*)d_in[4];
  const float* nn_w1   = (const float*)d_in[5];
  const float* nn_b1   = (const float*)d_in[6];
  const float* nn_w2   = (const float*)d_in[7];
  const float* nn_b2   = (const float*)d_in[8];
  const float* lin2_w  = (const float*)d_in[9];
  const float* lin2_b  = (const float*)d_in[10];
  float* out = (float*)d_out;

  // workspace layout (bytes)
  char* ws = (char*)d_ws;
  unsigned short* lin1b = (unsigned short*)(ws + 0);            // 128*128*2 = 32768
  unsigned short* w1b   = (unsigned short*)(ws + 32768);        // 128*64*2  = 16384
  unsigned short* w2b   = (unsigned short*)(ws + 49152);        // 32768
  unsigned short* lin2b = (unsigned short*)(ws + 81920);        // 32768
  float* h1  = (float*)(ws + 114688);                           // 50000*128*4 = 25.6 MB
  float* agg = (float*)(ws + 114688 + (size_t)N_NODES * CH * 4);// 25.6 MB

  hipMemsetAsync(agg, 0, (size_t)N_NODES * CH * sizeof(float), stream);

  k_convert<<<(CH * CH + 255) / 256, 256, 0, stream>>>(
      lin1_w, nn_w1, nn_w2, lin2_w, lin1b, w1b, w2b, lin2b);

  int gemm_blocks = (N_NODES + 63) / 64;  // 4 waves/block, 16 rows/wave
  k_node_gemm<<<gemm_blocks, 128, 0, stream>>>(h, lin1b, h1);

  k_edge<<<N_EDGES / 64, 128, 0, stream>>>(
      eattr, ew, eidx, w1b, nn_b1, w2b, nn_b2, h1, agg);

  k_out_gemm<<<gemm_blocks, 128, 0, stream>>>(h1, agg, lin2b, lin2_b, out);
}
